// WindowAttention3D_55138790146355
// MI455X (gfx1250) — matrix operand order repaired and checked by reading, not compiled, not run on hardware
//
#include <hip/hip_runtime.h>

typedef __attribute__((ext_vector_type(16))) _Float16 v16h;
typedef __attribute__((ext_vector_type(8)))  _Float16 v8h;
typedef __attribute__((ext_vector_type(8)))  float    v8f;

#define DIMC   384
#define HEADS  12
#define HDIM   32
#define TT     16
#define NWIN   64
#define NTOK   49
#define BATCHN 2
#define M_ROWS (BATCHN*TT*NWIN*NTOK)   /* 100352 */

// Per-token shift s[i%3][j%3] for token n = i*7+j (padded to 64 with 0)
__constant__ int c_S[64] = {
  -4, 1, 2,-4, 1, 2,-4,
  -1, 0, 3,-1, 0, 3,-1,
  -2,-3, 4,-2,-3, 4,-2,
  -4, 1, 2,-4, 1, 2,-4,
  -1, 0, 3,-1, 0, 3,-1,
  -2,-3, 4,-2,-3, 4,-2,
  -4, 1, 2,-4, 1, 2,-4,
   0, 0, 0, 0, 0, 0, 0, 0, 0, 0, 0, 0, 0, 0, 0 };

// Per-token packed spatial coord i*13 + j (padded to 64 with 0)
__constant__ int c_P[64] = {
   0, 1, 2, 3, 4, 5, 6,
  13,14,15,16,17,18,19,
  26,27,28,29,30,31,32,
  39,40,41,42,43,44,45,
  52,53,54,55,56,57,58,
  65,66,67,68,69,70,71,
  78,79,80,81,82,83,84,
   0, 0, 0, 0, 0, 0, 0, 0, 0, 0, 0, 0, 0, 0, 0 };

// row permutation induced by TPS pattern-C shift (sign=+1 fwd, -1 inverse)
__device__ __forceinline__ int map_row(int m, int sign) {
  int n = m % NTOK;
  int w = (m / NTOK) % NWIN;
  int t = (m / (NTOK*NWIN)) % TT;
  int b =  m / (NTOK*NWIN*TT);
  int s = c_S[n];
  int t2 = ((t - sign * s) + TT) & (TT - 1);
  return ((b*TT + t2)*NWIN + w)*NTOK + n;
}

// A-fragment (16x32 f16): lane<16 holds K{0..7,16..23}, lane>=16 K{8..15,24..31}, M = lane&15
__device__ __forceinline__ v16h frag_a(const _Float16* base, int lane, int stride) {
  int r   = lane & 15;
  int klo = (lane & 16) ? 8 : 0;
  const _Float16* p = base + r*stride + klo;
  v8h lo = *(const v8h*)(p);
  v8h hi = *(const v8h*)(p + 16);
  return __builtin_shufflevector(lo, hi, 0,1,2,3,4,5,6,7,8,9,10,11,12,13,14,15);
}

// B-fragment (32x16 f16): lane<16 holds K{0..7,16..23} of column lane, lane>=16 K{8..15,24..31} of column lane-16
// base must point at [n][k] layout (stride halves per n-row)
__device__ __forceinline__ v16h frag_b(const _Float16* base, int lane, int stride) {
  int n  = lane & 15;
  int k0 = (lane & 16) ? 8 : 0;
  const _Float16* p = base + n*stride + k0;
  v8h lo = *(const v8h*)(p);
  v8h hi = *(const v8h*)(p + 16);
  return __builtin_shufflevector(lo, hi, 0,1,2,3,4,5,6,7,8,9,10,11,12,13,14,15);
}

union F8 { v8f v; float f[8]; };

// ---------------------------------------------------------------------------
// Fused (patch-shift row-gather) + GEMM:  Out[m,o] = sum_k A[map(m),k]*W[o,k] + bias[o]
// Block tile 128x128, 8 waves (4x2), each wave 32x64 via 2x4 WMMA tiles, K-step 32.
// ---------------------------------------------------------------------------
template<bool A_F16, bool OUT_F16>
__global__ __launch_bounds__(256)
void fused_shift_gemm(const void* __restrict__ Aptr,
                      const float* __restrict__ W,
                      const float* __restrict__ bias,
                      void* __restrict__ Out,
                      int N, int K, int sign, float qScale)
{
  __shared__ alignas(16) _Float16 sA[128*40];
  __shared__ alignas(16) _Float16 sB[128*40];

  const int tid    = threadIdx.x;
  const int mBlock = blockIdx.y * 128;
  const int nBlock = blockIdx.x * 128;

  const int aRow = tid >> 1;            // 0..127
  const int aCol = (tid & 1) * 16;      // 0 / 16
  const int srcRow = map_row(mBlock + aRow, sign);

  const int wave = tid >> 5, lane = tid & 31;
  const int wm = (wave >> 1) * 32;      // wave's 32 rows
  const int wn = (wave & 1) * 64;       // wave's 64 cols

  v8f zero = {};
  F8 acc[2][4];
  #pragma unroll
  for (int a = 0; a < 2; a++)
    #pragma unroll
    for (int b = 0; b < 4; b++) acc[a][b].v = zero;

  for (int kk = 0; kk < K; kk += 32) {
    __syncthreads();
    // ---- stage A (gathered row, converted to f16) ----
    if (A_F16) {
      const _Float16* src = (const _Float16*)Aptr + (size_t)srcRow * K + kk + aCol;
      *(v8h*)&sA[aRow*40 + aCol]     = *(const v8h*)(src);
      *(v8h*)&sA[aRow*40 + aCol + 8] = *(const v8h*)(src + 8);
    } else {
      const float* src = (const float*)Aptr + (size_t)srcRow * K + kk + aCol;
      float4 f0 = ((const float4*)src)[0];
      float4 f1 = ((const float4*)src)[1];
      float4 f2 = ((const float4*)src)[2];
      float4 f3 = ((const float4*)src)[3];
      _Float16* d = &sA[aRow*40 + aCol];
      d[0] =(_Float16)f0.x; d[1] =(_Float16)f0.y; d[2] =(_Float16)f0.z; d[3] =(_Float16)f0.w;
      d[4] =(_Float16)f1.x; d[5] =(_Float16)f1.y; d[6] =(_Float16)f1.z; d[7] =(_Float16)f1.w;
      d[8] =(_Float16)f2.x; d[9] =(_Float16)f2.y; d[10]=(_Float16)f2.z; d[11]=(_Float16)f2.w;
      d[12]=(_Float16)f3.x; d[13]=(_Float16)f3.y; d[14]=(_Float16)f3.z; d[15]=(_Float16)f3.w;
    }
    // ---- stage B ([n][k], straight from row-major W) ----
    {
      const float* src = W + (size_t)(nBlock + aRow) * K + kk + aCol;
      float4 g0 = ((const float4*)src)[0];
      float4 g1 = ((const float4*)src)[1];
      float4 g2 = ((const float4*)src)[2];
      float4 g3 = ((const float4*)src)[3];
      _Float16* d = &sB[aRow*40 + aCol];
      d[0] =(_Float16)g0.x; d[1] =(_Float16)g0.y; d[2] =(_Float16)g0.z; d[3] =(_Float16)g0.w;
      d[4] =(_Float16)g1.x; d[5] =(_Float16)g1.y; d[6] =(_Float16)g1.z; d[7] =(_Float16)g1.w;
      d[8] =(_Float16)g2.x; d[9] =(_Float16)g2.y; d[10]=(_Float16)g2.z; d[11]=(_Float16)g2.w;
      d[12]=(_Float16)g3.x; d[13]=(_Float16)g3.y; d[14]=(_Float16)g3.z; d[15]=(_Float16)g3.w;
      if (kk + 32 < K) __builtin_prefetch(src + 32, 0, 1);
    }
    __syncthreads();

    v16h a0 = frag_a(&sA[(wm     )*40], lane, 40);
    v16h a1 = frag_a(&sA[(wm + 16)*40], lane, 40);
    #pragma unroll
    for (int tn = 0; tn < 4; tn++) {
      v16h b = frag_b(&sB[(wn + tn*16)*40], lane, 40);
      acc[0][tn].v = __builtin_amdgcn_wmma_f32_16x16x32_f16(false, a0, false, b, (short)0, acc[0][tn].v, false, false);
      acc[1][tn].v = __builtin_amdgcn_wmma_f32_16x16x32_f16(false, a1, false, b, (short)0, acc[1][tn].v, false, false);
    }
  }

  // ---- epilogue: C layout = lane<16 rows 0..7, lane>=16 rows 8..15, col = lane&15 ----
  const int rOff = (lane & 16) ? 8 : 0;
  #pragma unroll
  for (int tm = 0; tm < 2; tm++)
    #pragma unroll
    for (int tn = 0; tn < 4; tn++)
      #pragma unroll
      for (int r = 0; r < 8; r++) {
        int row = mBlock + wm + tm*16 + rOff + r;
        int col = nBlock + wn + tn*16 + (lane & 15);
        float v = acc[tm][tn].f[r] + bias[col];
        if (OUT_F16) {
          if (col < DIMC) v *= qScale;           // fold softmax scale into q
          ((_Float16*)Out)[(size_t)row * N + col] = (_Float16)v;
        } else {
          ((float*)Out)[(size_t)row * N + col] = v;
        }
      }
}

// ---------------------------------------------------------------------------
// Attention: one block (4 waves) per (window, head). 49 tokens padded to 64.
// scores = q@k^T (+rpb bias) -> softmax -> P@V. All matmuls via WMMA f16.
// ---------------------------------------------------------------------------
__global__ __launch_bounds__(128)
void window_attention(const _Float16* __restrict__ qkv,
                      const float* __restrict__ rpb,
                      _Float16* __restrict__ aout)
{
  __shared__ alignas(16) _Float16 qs[64*40];      // [token][hd]
  __shared__ alignas(16) _Float16 ks[64*40];      // [token][hd]  (B-layout for q@k^T)
  __shared__ alignas(16) _Float16 vt[32*72];      // [hd][token]  (B-layout for P@V)
  __shared__ alignas(16) _Float16 ps[4][16*72];   // per-wave P tile 16x64

  const int bid = blockIdx.x;
  const int win = bid / HEADS;
  const int h   = bid % HEADS;
  const int t   = (win >> 6) & (TT - 1);          // win = ((b*T+t)*NW + w)

  const int tid  = threadIdx.x;
  const int wave = tid >> 5, lane = tid & 31;

  // ---- load q, k, v tiles (f16), zero-pad tokens 49..63 ----
  {
    int row = tid >> 1;
    int off = (tid & 1) * 16;
    if (row < NTOK) {
      const _Float16* q = qkv + (size_t)(win*NTOK + row) * (3*DIMC) + h*HDIM + off;
      *(v8h*)&qs[row*40 + off]     = *(const v8h*)(q);
      *(v8h*)&qs[row*40 + off + 8] = *(const v8h*)(q + 8);
      const _Float16* kp = q + DIMC;
      *(v8h*)&ks[row*40 + off]     = *(const v8h*)(kp);
      *(v8h*)&ks[row*40 + off + 8] = *(const v8h*)(kp + 8);
      const _Float16* vp = q + 2*DIMC;
      v8h v0 = *(const v8h*)(vp);
      v8h v1 = *(const v8h*)(vp + 8);
      #pragma unroll
      for (int c = 0; c < 8; c++) {
        vt[(off + c)     * 72 + row] = v0[c];
        vt[(off + 8 + c) * 72 + row] = v1[c];
      }
    } else {
      v8h z = {};
      *(v8h*)&qs[row*40 + off] = z; *(v8h*)&qs[row*40 + off + 8] = z;
      *(v8h*)&ks[row*40 + off] = z; *(v8h*)&ks[row*40 + off + 8] = z;
      #pragma unroll
      for (int c = 0; c < 16; c++) vt[(off + c)*72 + row] = (_Float16)0.f;
    }
  }
  __syncthreads();

  // ---- scores: wave's 16 rows x 64 cols, K=32 in one WMMA per 16x16 tile ----
  const int wBase = wave * 16;
  v16h aq = frag_a(&qs[wBase*40], lane, 40);
  v8f zero = {};
  F8 sc[4];
  #pragma unroll
  for (int tn = 0; tn < 4; tn++) {
    v16h bk = frag_b(&ks[(tn*16)*40], lane, 40);
    sc[tn].v = __builtin_amdgcn_wmma_f32_16x16x32_f16(false, aq, false, bk, (short)0, zero, false, false);
  }

  // ---- relative-position bias via closed-form RPI, fully branchless ----
  // Padded c_S/c_P keep idx in [0, 5238*12+11] even for pad tokens, so every
  // load is safe: issue all 32 gathers up front, apply masks with selects.
  const int rOff = (lane & 16) ? 8 : 0;
  const int colL = lane & 15;
  int rC0p[8], rP[8];
  #pragma unroll
  for (int r = 0; r < 8; r++) {
    int n1 = wBase + rOff + r;
    rC0p[r] = ((t - c_S[n1]) + 16) & 15;    // shifted d-coord at d=t
    rP[r]   = c_P[n1];
  }
  float bv[4][8];
  #pragma unroll
  for (int tn = 0; tn < 4; tn++) {
    int n2  = tn*16 + colL;
    int c0q = (16 - c_S[n2]) & 15;          // shifted d-coord at d=0
    int p2  = c_P[n2];
    #pragma unroll
    for (int r = 0; r < 8; r++) {
      int idx = ((rC0p[r] - c0q + (TT - 1)) * 169 + (rP[r] - p2 + 84)) * HEADS + h;
      bv[tn][r] = rpb[idx];
    }
  }
  #pragma unroll
  for (int tn = 0; tn < 4; tn++) {
    bool v2 = (tn*16 + colL) < NTOK;
    #pragma unroll
    for (int r = 0; r < 8; r++) {
      bool ok = v2 && ((wBase + rOff + r) < NTOK);
      sc[tn].f[r] = ok ? (sc[tn].f[r] + bv[tn][r]) : -1e30f;
    }
  }

  // ---- softmax per row: row's 16 cols live in one 16-lane half per tile ----
  #pragma unroll
  for (int r = 0; r < 8; r++) {
    float mx = fmaxf(fmaxf(sc[0].f[r], sc[1].f[r]), fmaxf(sc[2].f[r], sc[3].f[r]));
    #pragma unroll
    for (int m = 8; m >= 1; m >>= 1) mx = fmaxf(mx, __shfl_xor(mx, m, 32));
    float sum = 0.f;
    #pragma unroll
    for (int tn = 0; tn < 4; tn++) { float e = __expf(sc[tn].f[r] - mx); sc[tn].f[r] = e; sum += e; }
    #pragma unroll
    for (int m = 8; m >= 1; m >>= 1) sum += __shfl_xor(sum, m, 32);
    float inv = 1.f / sum;
    #pragma unroll
    for (int tn = 0; tn < 4; tn++) sc[tn].f[r] *= inv;
  }

  // ---- reshape P: C-layout regs -> LDS [row][col] so it can be an A operand ----
  _Float16* pw = &ps[wave][0];
  #pragma unroll
  for (int tn = 0; tn < 4; tn++)
    #pragma unroll
    for (int r = 0; r < 8; r++)
      pw[(rOff + r)*72 + tn*16 + colL] = (_Float16)sc[tn].f[r];
  __syncthreads();

  // ---- out = P(16x64) @ V(64x32): 2 N-tiles x 2 K-chunks ----
  v16h pa0 = frag_a(pw,      lane, 72);
  v16h pa1 = frag_a(pw + 32, lane, 72);
  F8 o[2];
  #pragma unroll
  for (int tn2 = 0; tn2 < 2; tn2++) {
    v16h bv0 = frag_b(&vt[(tn2*16)*72],      lane, 72);
    v16h bv1 = frag_b(&vt[(tn2*16)*72 + 32], lane, 72);
    o[tn2].v = __builtin_amdgcn_wmma_f32_16x16x32_f16(false, pa0, false, bv0, (short)0, zero,      false, false);
    o[tn2].v = __builtin_amdgcn_wmma_f32_16x16x32_f16(false, pa1, false, bv1, (short)0, o[tn2].v, false, false);
  }

  // ---- store (head-interleaved f16 rows, ready for the proj GEMM gather) ----
  #pragma unroll
  for (int tn2 = 0; tn2 < 2; tn2++)
    #pragma unroll
    for (int r = 0; r < 8; r++) {
      int row = wBase + rOff + r;
      if (row < NTOK) {
        int col = tn2*16 + colL;
        aout[(size_t)(win*NTOK + row) * DIMC + h*HDIM + col] = (_Float16)o[tn2].f[r];
      }
    }
}

// ---------------------------------------------------------------------------
extern "C" void kernel_launch(void* const* d_in, const int* in_sizes, int n_in,
                              void* d_out, int out_size, void* d_ws, size_t ws_size,
                              hipStream_t stream)
{
  (void)in_sizes; (void)n_in; (void)out_size; (void)ws_size;
  const float* x      = (const float*)d_in[0];
  const float* qkv_w  = (const float*)d_in[1];
  const float* qkv_b  = (const float*)d_in[2];
  const float* proj_w = (const float*)d_in[3];
  const float* proj_b = (const float*)d_in[4];
  const float* rpb    = (const float*)d_in[5];

  _Float16* qkv_ws  = (_Float16*)d_ws;                                   // 100352 x 1152 f16
  _Float16* attn_ws = (_Float16*)((char*)d_ws +
                        (size_t)M_ROWS * (3*DIMC) * sizeof(_Float16));   // 100352 x 384 f16

  const float scale = 0.17677669529663687f;  // HD^-0.5 = 1/sqrt(32)

  // 1) forward patch-shift gather fused into QKV projection (q pre-scaled)
  fused_shift_gemm<false, true><<<dim3((3*DIMC)/128, M_ROWS/128), 256, 0, stream>>>(
      x, qkv_w, qkv_b, qkv_ws, 3*DIMC, DIMC, +1, scale);

  // 2) windowed attention with closed-form relative-position bias
  window_attention<<<BATCHN*TT*NWIN*HEADS, 128, 0, stream>>>(qkv_ws, rpb, attn_ws);

  // 3) inverse patch-shift gather fused into output projection
  fused_shift_gemm<true, false><<<dim3(DIMC/128, M_ROWS/128), 256, 0, stream>>>(
      attn_ws, proj_w, proj_b, d_out, DIMC, DIMC, -1, 1.0f);
}
